// GETSICSoftmaxLayer_49855980372379
// MI455X (gfx1250) — matrix operand order repaired and checked by reading, not compiled, not run on hardware
//
#include <hip/hip_runtime.h>
#include <math.h>

#define DIMN   64
#define NHEADS 4
#define TILE   16
#define GAMMAF 1.0f
#define SICS   0.5f

typedef __attribute__((ext_vector_type(16))) _Float16 v16h;
typedef __attribute__((ext_vector_type(8)))  float    v8f;

// ---- WMMA fragment loaders (layouts per CDNA5 ISA 7.12.2, wave32) ----
// A (16x32 f16): lane L holds M = mt*16 + (L&15); k(j) = kc*32 + (j/8)*16 + ((L>>4)&1)*8 + (j%8)
__device__ __forceinline__ v16h ldsA(const _Float16* W, int mt, int kc, int lane) {
  const _Float16* p = W + (mt * 16 + (lane & 15)) * DIMN + kc * 32 + ((lane >> 4) & 1) * 8;
  v16h a;
#pragma unroll
  for (int j = 0; j < 8; ++j) { a[j] = p[j]; a[j + 8] = p[16 + j]; }
  return a;
}
// B (32x16 f16): lane L holds N = (L&15); k(j) = kc*32 + (j/8)*16 + ((L>>4)&1)*8 + (j%8)
__device__ __forceinline__ v16h ldsB(const _Float16* X, int kc, int lane) {
  const _Float16* p = X + (lane & 15) * DIMN + kc * 32 + ((lane >> 4) & 1) * 8;
  v16h b;
#pragma unroll
  for (int j = 0; j < 8; ++j) { b[j] = p[j]; b[j + 8] = p[16 + j]; }
  return b;
}

// complex (Wr+iWi) * (xr+ixi): acc_re += Wr*xr + Wi*(-xi); acc_im += Wr*xi + Wi*xr
__device__ __forceinline__ void cplx_wmma(const _Float16* Wr, const _Float16* Wi,
                                          const _Float16* Xr, const _Float16* Xi,
                                          const _Float16* Xni, int mt, int lane,
                                          v8f& are, v8f& aim) {
#pragma unroll
  for (int kc = 0; kc < 2; ++kc) {
    v16h ar = ldsA(Wr, mt, kc, lane);
    v16h ai = ldsA(Wi, mt, kc, lane);
    v16h br = ldsB(Xr, kc, lane);
    v16h bi = ldsB(Xi, kc, lane);
    v16h bn = ldsB(Xni, kc, lane);
    are = __builtin_amdgcn_wmma_f32_16x16x32_f16(false, ar, false, br, (short)0, are, false, false);
    are = __builtin_amdgcn_wmma_f32_16x16x32_f16(false, ai, false, bn, (short)0, are, false, false);
    aim = __builtin_amdgcn_wmma_f32_16x16x32_f16(false, ar, false, bi, (short)0, aim, false, false);
    aim = __builtin_amdgcn_wmma_f32_16x16x32_f16(false, ai, false, br, (short)0, aim, false, false);
  }
}

// ---------------- K1: attention logits + segment max + sign gate ----------------
__global__ __launch_bounds__(128) void k1_logits(
    const float* __restrict__ h_r, const float* __restrict__ h_i,
    const float* __restrict__ W_r, const float* __restrict__ W_i,
    const float* __restrict__ Q_r, const float* __restrict__ Q_i,
    const int* __restrict__ src, const int* __restrict__ dst, int nE,
    float* __restrict__ logits, float* __restrict__ sgate,
    unsigned int* __restrict__ segmax) {
  __shared__ _Float16 sW[NHEADS][2][DIMN * DIMN];
  __shared__ _Float16 sQ[NHEADS][2][DIMN * DIMN];
  __shared__ _Float16 sXs[3][TILE * DIMN];   // hs: re, im, -im  (B layout = per-edge rows)
  __shared__ _Float16 sXd[3][TILE * DIMN];   // hd: re, im, -im
  __shared__ float sHr[TILE * DIMN];
  __shared__ float sHi[TILE * DIMN];
  __shared__ float sN2[TILE];
  __shared__ float sSim[TILE];

  const int tid  = threadIdx.x;
  const int wave = tid >> 5;     // = head
  const int lane = tid & 31;

  for (int i = tid; i < NHEADS * DIMN * DIMN; i += blockDim.x) {
    int h = i / (DIMN * DIMN), r = i % (DIMN * DIMN);
    sW[h][0][r] = (_Float16)W_r[i];
    sW[h][1][r] = (_Float16)W_i[i];
    sQ[h][0][r] = (_Float16)Q_r[i];
    sQ[h][1][r] = (_Float16)Q_i[i];
  }

  const int nTiles = (nE + TILE - 1) / TILE;
  for (int tile = blockIdx.x; tile < nTiles; tile += gridDim.x) {
    __syncthreads();
    if (tid < TILE) { sN2[tid] = 0.f; sSim[tid] = 0.f; }
    __syncthreads();
    const int e0 = tile * TILE;
    for (int i = tid; i < TILE * DIMN; i += blockDim.x) {
      int le = i >> 6, d = i & 63;
      int e = e0 + le;
      float sr = 0.f, si = 0.f, tr = 0.f, ti = 0.f;
      if (e < nE) {
        int a = src[e], b = dst[e];
        sr = h_r[a * DIMN + d]; si = h_i[a * DIMN + d];
        tr = h_r[b * DIMN + d]; ti = h_i[b * DIMN + d];
      }
      sXs[0][i] = (_Float16)sr; sXs[1][i] = (_Float16)si; sXs[2][i] = (_Float16)(-si);
      sXd[0][i] = (_Float16)tr; sXd[1][i] = (_Float16)ti; sXd[2][i] = (_Float16)(-ti);
      sHr[i] = tr; sHi[i] = ti;
      atomicAdd(&sN2[le], tr * tr + ti * ti);
      atomicAdd(&sSim[le], sr * tr);
    }
    __syncthreads();

    // Wh = W[head] * hs   (4 M-tiles of 16 rows)
    v8f wre[4], wim[4];
#pragma unroll
    for (int mt = 0; mt < 4; ++mt) {
      v8f z = {0.f, 0.f, 0.f, 0.f, 0.f, 0.f, 0.f, 0.f};
      wre[mt] = z; wim[mt] = z;
      cplx_wmma(sW[wave][0], sW[wave][1], sXs[0], sXs[1], sXs[2], mt, lane, wre[mt], wim[mt]);
    }
    const int n  = lane & 15;          // edge column
    const int mh = (lane >> 4) * 8;    // C-layout row offset
    float dre = 0.f, dimg = 0.f;       // dot = sum conj(hd)*Wh
#pragma unroll
    for (int mt = 0; mt < 4; ++mt) {
#pragma unroll
      for (int v = 0; v < 8; ++v) {
        int d = mt * 16 + mh + v;
        float hr = sHr[n * DIMN + d], hi = sHi[n * DIMN + d];
        dre  += hr * wre[mt][v] + hi * wim[mt][v];
        dimg += hr * wim[mt][v] - hi * wre[mt][v];
      }
    }
    dre  += __shfl_xor(dre, 16);
    dimg += __shfl_xor(dimg, 16);
    float n2  = fmaxf(sN2[n], 1e-6f);
    float cre = dre / n2, cim = dimg / n2;

    // s = sum conj(Qh) * r,  r = Wh - SICS * hd * c
    float sre = 0.f, simg = 0.f;
#pragma unroll
    for (int mt = 0; mt < 4; ++mt) {
      v8f z = {0.f, 0.f, 0.f, 0.f, 0.f, 0.f, 0.f, 0.f};
      v8f qre = z, qim = z;
      cplx_wmma(sQ[wave][0], sQ[wave][1], sXd[0], sXd[1], sXd[2], mt, lane, qre, qim);
#pragma unroll
      for (int v = 0; v < 8; ++v) {
        int d = mt * 16 + mh + v;
        float hr = sHr[n * DIMN + d], hi = sHi[n * DIMN + d];
        float rre = wre[mt][v] - SICS * (hr * cre - hi * cim);
        float rim = wim[mt][v] - SICS * (hr * cim + hi * cre);
        sre  += qre[v] * rre + qim[v] * rim;
        simg += qre[v] * rim - qim[v] * rre;
      }
    }
    sre  += __shfl_xor(sre, 16);
    simg += __shfl_xor(simg, 16);

    if (lane < 16) {
      int e = e0 + n;
      if (e < nE) {
        float lg = GAMMAF * sqrtf(sre * sre + simg * simg) * 0.125f;
        if (!isfinite(lg)) lg = 0.f;
        logits[wave * nE + e] = lg;
        atomicMax(&segmax[dst[e] * NHEADS + wave], __float_as_uint(lg));  // lg >= 0
        if (wave == 0) {
          float sm = sSim[n];
          sgate[e] = (sm > 0.f) ? 1.f : ((sm < 0.f) ? 0.f : 0.5f);
        }
      }
    }
  }
}

// ---------------- K2: exp(logit - segmax), segment sum ----------------
__global__ __launch_bounds__(256) void k2_expsum(
    const float* __restrict__ logits, const int* __restrict__ dst, int nE,
    const unsigned int* __restrict__ segmax, float* __restrict__ ex,
    float* __restrict__ segsum) {
  int i = blockIdx.x * blockDim.x + threadIdx.x;
  if (i < NHEADS * nE) {
    int h = i / nE, e = i % nE;
    int t = dst[e];
    float m = __uint_as_float(segmax[t * NHEADS + h]);
    float v = expf(logits[i] - m);
    ex[i] = v;
    atomicAdd(&segsum[t * NHEADS + h], v);
  }
}

// ---------------- K3: messages, scatter-add to nodes ----------------
__global__ __launch_bounds__(128) void k3_msg(
    const float* __restrict__ h_r, const float* __restrict__ h_i,
    const float* __restrict__ W_r, const float* __restrict__ W_i,
    const int* __restrict__ src, const int* __restrict__ dst, int nE,
    const float* __restrict__ ex, const float* __restrict__ segsum,
    const float* __restrict__ sgate, const float* __restrict__ gate_raw,
    float* __restrict__ upd) {
  __shared__ _Float16 sW[NHEADS][2][DIMN * DIMN];
  __shared__ _Float16 sXs[3][TILE * DIMN];
  __shared__ float sHr[TILE * DIMN];
  __shared__ float sHi[TILE * DIMN];
  __shared__ float sN2[TILE];
  __shared__ float sAcc[TILE * DIMN * 2];   // [edge][d][re/im]
  __shared__ int   sDst[TILE];

  const int tid  = threadIdx.x;
  const int wave = tid >> 5;
  const int lane = tid & 31;
  const float g  = 1.f / (1.f + expf(-gate_raw[0]));
  const float gh = g * SICS;   // base = Wh - gate*SICS*proj

  for (int i = tid; i < NHEADS * DIMN * DIMN; i += blockDim.x) {
    int h = i / (DIMN * DIMN), r = i % (DIMN * DIMN);
    sW[h][0][r] = (_Float16)W_r[i];
    sW[h][1][r] = (_Float16)W_i[i];
  }

  const int nTiles = (nE + TILE - 1) / TILE;
  for (int tile = blockIdx.x; tile < nTiles; tile += gridDim.x) {
    __syncthreads();
    if (tid < TILE) { sN2[tid] = 0.f; sDst[tid] = 0; }
    for (int i = tid; i < TILE * DIMN * 2; i += blockDim.x) sAcc[i] = 0.f;
    __syncthreads();
    const int e0 = tile * TILE;
    for (int i = tid; i < TILE * DIMN; i += blockDim.x) {
      int le = i >> 6, d = i & 63;
      int e = e0 + le;
      float sr = 0.f, si = 0.f, tr = 0.f, ti = 0.f;
      if (e < nE) {
        int a = src[e], b = dst[e];
        sr = h_r[a * DIMN + d]; si = h_i[a * DIMN + d];
        tr = h_r[b * DIMN + d]; ti = h_i[b * DIMN + d];
        if (d == 0) sDst[le] = b;
      }
      sXs[0][i] = (_Float16)sr; sXs[1][i] = (_Float16)si; sXs[2][i] = (_Float16)(-si);
      sHr[i] = tr; sHi[i] = ti;
      atomicAdd(&sN2[le], tr * tr + ti * ti);
    }
    __syncthreads();

    v8f wre[4], wim[4];
#pragma unroll
    for (int mt = 0; mt < 4; ++mt) {
      v8f z = {0.f, 0.f, 0.f, 0.f, 0.f, 0.f, 0.f, 0.f};
      wre[mt] = z; wim[mt] = z;
      cplx_wmma(sW[wave][0], sW[wave][1], sXs[0], sXs[1], sXs[2], mt, lane, wre[mt], wim[mt]);
    }
    const int n  = lane & 15;
    const int mh = (lane >> 4) * 8;
    float dre = 0.f, dimg = 0.f;
#pragma unroll
    for (int mt = 0; mt < 4; ++mt) {
#pragma unroll
      for (int v = 0; v < 8; ++v) {
        int d = mt * 16 + mh + v;
        float hr = sHr[n * DIMN + d], hi = sHi[n * DIMN + d];
        dre  += hr * wre[mt][v] + hi * wim[mt][v];
        dimg += hr * wim[mt][v] - hi * wre[mt][v];
      }
    }
    dre  += __shfl_xor(dre, 16);
    dimg += __shfl_xor(dimg, 16);
    float n2  = fmaxf(sN2[n], 1e-6f);
    float cre = dre / n2, cim = dimg / n2;

    int   e   = e0 + n;
    float asg = 0.f;
    if (e < nE) {
      int t = sDst[n];
      float a = ex[wave * nE + e] / (segsum[t * NHEADS + wave] + 1e-16f);
      asg = a * sgate[e];
      if (!isfinite(asg)) asg = 0.f;
    }
#pragma unroll
    for (int mt = 0; mt < 4; ++mt) {
#pragma unroll
      for (int v = 0; v < 8; ++v) {
        int d = mt * 16 + mh + v;
        float hr = sHr[n * DIMN + d], hi = sHi[n * DIMN + d];
        float bre = wre[mt][v] - gh * (hr * cre - hi * cim);
        float bim = wim[mt][v] - gh * (hr * cim + hi * cre);
        atomicAdd(&sAcc[n * (DIMN * 2) + d * 2 + 0], asg * bre);
        atomicAdd(&sAcc[n * (DIMN * 2) + d * 2 + 1], asg * bim);
      }
    }
    __syncthreads();
    for (int i = tid; i < TILE * DIMN * 2; i += blockDim.x) {
      int le = i >> 7;
      if (e0 + le < nE) atomicAdd(&upd[(size_t)sDst[le] * (DIMN * 2) + (i & 127)], sAcc[i]);
    }
  }
}

// ---------------- K4: residual + NodeNorm (ddof=1) + ModReLU ----------------
__global__ __launch_bounds__(256) void k4_final(
    const float* __restrict__ h_r, const float* __restrict__ h_i,
    const float* __restrict__ upd, const float* __restrict__ mb,
    float* __restrict__ out, int nN) {
  int gw   = (blockIdx.x * blockDim.x + threadIdx.x) >> 5;  // node = global wave id
  int lane = threadIdx.x & 31;
  if (gw >= nN) return;
  const int base = gw * DIMN;
  float re[2], im[2];
#pragma unroll
  for (int p = 0; p < 2; ++p) {
    int d = lane + p * 32;
    re[p] = h_r[base + d] + upd[(size_t)(base + d) * 2 + 0];
    im[p] = h_i[base + d] + upd[(size_t)(base + d) * 2 + 1];
  }
  float sr = re[0] + re[1], si = im[0] + im[1];
#pragma unroll
  for (int o = 16; o > 0; o >>= 1) { sr += __shfl_xor(sr, o); si += __shfl_xor(si, o); }
  float mr = sr * (1.f / 64.f), mi = si * (1.f / 64.f);
  float vr = (re[0] - mr) * (re[0] - mr) + (re[1] - mr) * (re[1] - mr);
  float vi = (im[0] - mi) * (im[0] - mi) + (im[1] - mi) * (im[1] - mi);
#pragma unroll
  for (int o = 16; o > 0; o >>= 1) { vr += __shfl_xor(vr, o); vi += __shfl_xor(vi, o); }
  float stdr = fmaxf(sqrtf(vr * (1.f / 63.f)), 1e-5f);
  float stdi = fmaxf(sqrtf(vi * (1.f / 63.f)), 1e-5f);
#pragma unroll
  for (int p = 0; p < 2; ++p) {
    int d = lane + p * 32;
    float xr = (re[p] - mr) / stdr;
    float xi = (im[p] - mi) / stdi;
    float mag = fmaxf(sqrtf(xr * xr + xi * xi), 1e-6f);
    float gt  = fmaxf(mag + mb[d], 0.f);
    float sc  = gt / mag;
    out[(size_t)(base + d) * 2 + 0] = xr * sc;
    out[(size_t)(base + d) * 2 + 1] = xi * sc;
  }
}

extern "C" void kernel_launch(void* const* d_in, const int* in_sizes, int n_in,
                              void* d_out, int out_size, void* d_ws, size_t ws_size,
                              hipStream_t stream) {
  const float* h_r = (const float*)d_in[0];
  const float* h_i = (const float*)d_in[1];
  const float* W_r = (const float*)d_in[2];
  const float* W_i = (const float*)d_in[3];
  const float* Q_r = (const float*)d_in[4];
  const float* Q_i = (const float*)d_in[5];
  const float* grw = (const float*)d_in[6];
  const float* mb  = (const float*)d_in[7];
  const int*   src = (const int*)d_in[8];
  const int*   dst = (const int*)d_in[9];
  const int nE = in_sizes[8];
  const int nN = in_sizes[0] / DIMN;

  size_t off = 0;
  auto carve = [&](size_t bytes) {
    void* q = (char*)d_ws + off;
    off += (bytes + 255) & ~(size_t)255;
    return q;
  };
  float*        logits = (float*)carve((size_t)NHEADS * nE * 4);
  float*        ex     = (float*)carve((size_t)NHEADS * nE * 4);
  float*        sgate  = (float*)carve((size_t)nE * 4);
  unsigned int* segmax = (unsigned int*)carve((size_t)nN * NHEADS * 4);
  float*        segsum = (float*)carve((size_t)nN * NHEADS * 4);
  float*        upd    = (float*)carve((size_t)nN * DIMN * 2 * 4);

  hipMemsetAsync(segmax, 0, (size_t)nN * NHEADS * 4, stream);
  hipMemsetAsync(segsum, 0, (size_t)nN * NHEADS * 4, stream);
  hipMemsetAsync(upd,    0, (size_t)nN * DIMN * 2 * 4, stream);

  const int nTiles = (nE + TILE - 1) / TILE;
  const int g1 = nTiles < 960 ? nTiles : 960;

  k1_logits<<<g1, 128, 0, stream>>>(h_r, h_i, W_r, W_i, Q_r, Q_i, src, dst, nE,
                                    logits, sgate, segmax);
  const int tot = NHEADS * nE;
  k2_expsum<<<(tot + 255) / 256, 256, 0, stream>>>(logits, dst, nE, segmax, ex, segsum);
  k3_msg<<<g1, 128, 0, stream>>>(h_r, h_i, W_r, W_i, src, dst, nE, ex, segsum,
                                 sgate, grw, upd);
  k4_final<<<(nN + 7) / 8, 256, 0, stream>>>(h_r, h_i, upd, mb, (float*)d_out, nN);
}
